// SelfAtten_18408229830996
// MI455X (gfx1250) — matrix operand order repaired and checked by reading, not compiled, not run on hardware
//
#include <hip/hip_runtime.h>

// Self-attention forward: B=4, S=4096, D=512, fp32 in/out.
// bf16 WMMA (v_wmma_f32_16x16x32_bf16) for all 4 GEMMs, fp32 softmax.
// All WMMA B-operands arranged so each lane reads a contiguous 32B run.
// Scores row-block (16 x 4096 fp32 = 256KB) lives in CDNA5's 320KB LDS.
// Q-tile staged via TDM (tensor_load_to_lds, TENSORcnt); AO-tile staged via
// GLOBAL_LOAD_ASYNC_TO_LDS_B128 (ASYNCcnt).

#define S_LEN 4096
#define DMODEL 512
#define ND3 1536
#define ATTN_THREADS 512

typedef __attribute__((ext_vector_type(16))) __bf16 v16bf;
typedef __attribute__((ext_vector_type(8)))  float  v8f;
typedef __attribute__((ext_vector_type(4)))  unsigned int v4u;
typedef __attribute__((ext_vector_type(8)))  int v8i;
typedef __attribute__((ext_vector_type(4)))  int v4i;

// ---- CDNA5 async memory->LDS helpers ----------------------------------
// AS(3) pointers on amdgcn are 32-bit LDS byte offsets; ptrtoint gives the
// offset the async/TDM instructions' LDS-address operands want.
typedef __attribute__((address_space(3))) char lds_char;
__device__ __forceinline__ unsigned lds_offset(const void* p) {
  return (unsigned)(unsigned long long)(lds_char*)(p);
}
// Per-lane 16B copy: LDS[lds_off] = MEM[gaddr..gaddr+15]; tracked by ASYNCcnt.
__device__ __forceinline__ void async_copy16(unsigned lds_off, const void* g) {
  asm volatile("global_load_async_to_lds_b128 %0, %1, off"
               :: "v"(lds_off), "v"((unsigned long long)g)
               : "memory");
}
__device__ __forceinline__ void wait_async0() {
  asm volatile("s_wait_asynccnt 0x0" ::: "memory");
}

// ---- TDM: one-descriptor 2D tile DMA, global -> LDS (ISA ch.8 D#) ----
// 2D tile of 16-bit elements: tensor (tensor_d0 x tensor_d1), tile
// (tile_d0 x tile_d1), row stride stride0 (all in elements).
// This toolchain's builtin takes 6 args:
//   (uint32x4 g0, int32x8 g1, int32x4 g2, int32x4 g3, int32x8, i32 cpol)
__device__ __forceinline__ void tdm_load_2d_bf16(
    unsigned lds_addr, const void* gaddr, unsigned tensor_d0, unsigned tensor_d1,
    unsigned tile_d0, unsigned tile_d1, unsigned stride0) {
  const unsigned long long ga = (unsigned long long)gaddr;
  v4u g0;
  g0[0] = 1u;                                              // count=1 (valid D#)
  g0[1] = lds_addr;                                        // lds_addr (bytes)
  g0[2] = (unsigned)ga;                                    // global_addr[31:0]
  g0[3] = ((unsigned)(ga >> 32) & 0x01FFFFFFu) | 0x80000000u;  // [56:32]|type=2
  v8i g1;
  g1[0] = (int)(1u << 16);                                 // data_size=1 (2B)
  g1[1] = (int)(tensor_d0 << 16);                          // tensor_dim0 lo16
  g1[2] = (int)((tensor_d0 >> 16) | (tensor_d1 << 16));    // d0 hi | d1 lo
  g1[3] = (int)((tensor_d1 >> 16) | (tile_d0 << 16));      // d1 hi | tile_dim0
  g1[4] = (int)(tile_d1 & 0xFFFFu);                        // tile_dim1; dim2=0
  g1[5] = (int)stride0;                                    // dim0_stride lo32
  g1[6] = 0;                                               // stride hi, d1_stride
  g1[7] = 0;
  const v4i z4 = {0, 0, 0, 0};                             // 2D: groups 2/3 unused
  const v8i z8 = {0, 0, 0, 0, 0, 0, 0, 0};
  __builtin_amdgcn_tensor_load_to_lds(g0, g1, z4, z4, z8, 0);
}
__device__ __forceinline__ void wait_tensor0() {
  __builtin_amdgcn_s_wait_tensorcnt(0);
}

// ---- WMMA fragment helpers (wave32; A and B both use contiguous k = 16*(L>=16) + e) ----

// A-matrix 16x32 bf16: lane L holds row M=L&15; element e covers
// K = khalf + e, khalf=(L>=16)*16  => one 32B contiguous run.
__device__ __forceinline__ v16bf pack_a_bf16(const __bf16* base, int ld, int lane) {
  const int m = lane & 15;
  const int khalf = (lane >> 4) << 4;
  const __bf16* row = base + m * ld + khalf;
  v16bf a;
#pragma unroll
  for (int e = 0; e < 16; ++e)
    a[e] = row[e];
  return a;
}

__device__ __forceinline__ v16bf pack_a_f32(const float* base, int ld, int lane) {
  const int m = lane & 15;
  const int khalf = (lane >> 4) << 4;
  const float* row = base + m * ld + khalf;
  v16bf a;
#pragma unroll
  for (int e = 0; e < 16; ++e)
    a[e] = (__bf16)row[e];
  return a;
}

// B-matrix 32x16 bf16: lane L holds column N=L&15; element e -> K = (L>=16)*16+e.
// Caller passes this lane's pointer at (K=lane-half base, N); elements contiguous.
__device__ __forceinline__ v16bf pack_b_bf16(const __bf16* col) {
  v16bf b;
#pragma unroll
  for (int e = 0; e < 16; ++e) b[e] = col[e];
  return b;
}

__device__ __forceinline__ v8f wmma_bf16(v16bf a, v16bf b, v8f c) {
  // (neg_a, A, neg_b, B, c_mod, C, reuse_a, reuse_b)
  return __builtin_amdgcn_wmma_f32_16x16x32_bf16(false, a, false, b, (short)0, c,
                                                 false, false);
}

// ---- Kernel 0: transpose + convert weights to bf16 (one-time prep) ----
// WqkvT[n][k] = Wqkv[k][n]  (bf16, [ND3][DMODEL])
// WoT[n][k]   = Wo[k][n]    (bf16, [DMODEL][DMODEL])
__global__ __launch_bounds__(256) void prep_weights_kernel(
    const float* __restrict__ Wqkv, const float* __restrict__ Wo,
    __bf16* __restrict__ WqkvT, __bf16* __restrict__ WoT) {
  const int idx = blockIdx.x * 256 + threadIdx.x;
  if (idx < ND3 * DMODEL) {
    const int n = idx / DMODEL, k = idx & (DMODEL - 1);
    WqkvT[idx] = (__bf16)Wqkv[(size_t)k * ND3 + n];
  }
  if (idx < DMODEL * DMODEL) {
    const int n = idx / DMODEL, k = idx & (DMODEL - 1);
    WoT[idx] = (__bf16)Wo[(size_t)k * DMODEL + n];
  }
}

// ---- Kernel 1: QKV = x @ Wqkv + bqkv; scale folded into Q; V stored
// transposed as Vt[b][d][s] so the P@V B-fragments are lane-contiguous. ----
// grid: (B*S/16, ND3/(16*8)), block 256 = 8 waves, one 16x16 C-tile per wave.
__global__ __launch_bounds__(256) void qkv_proj_kernel(
    const float* __restrict__ x, const __bf16* __restrict__ WqkvT,
    const float* __restrict__ bqkv, __bf16* __restrict__ Q,
    __bf16* __restrict__ Kg, __bf16* __restrict__ Vt) {
  __shared__ __bf16 atile[16 * DMODEL];  // 16KB
  const int tid = threadIdx.x;
  const int m0 = blockIdx.x * 16;
  // fp32 -> bf16 conversion required, so stage through VGPRs.
  for (int i = tid; i < 16 * DMODEL; i += 256)
    atile[i] = (__bf16)x[(size_t)(m0 + (i >> 9)) * DMODEL + (i & (DMODEL - 1))];
  __syncthreads();

  const int wave = tid >> 5, lane = tid & 31;
  const int n0 = (blockIdx.y * 8 + wave) * 16;
  const int ncol = n0 + (lane & 15);
  const int kb = (lane >> 4) << 4;
  const int mh = (lane >> 4) << 3;
  const __bf16* bcol = WqkvT + (size_t)ncol * DMODEL + kb;  // contiguous in k

  v8f acc = {};
#pragma unroll 4
  for (int kc = 0; kc < DMODEL / 32; ++kc) {
    v16bf a = pack_a_bf16(atile + kc * 32, DMODEL, lane);
    v16bf b = pack_b_bf16(bcol + kc * 32);
    acc = wmma_bf16(a, b, acc);
  }

  const float bias = bqkv[ncol];
  const int region = ncol >> 9;  // 0=Q 1=K 2=V
  const int nl = ncol & (DMODEL - 1);
  if (region == 2) {
#pragma unroll
    for (int r = 0; r < 8; ++r) {
      const int m = m0 + mh + r;
      const int b = m >> 12, s = m & (S_LEN - 1);
      Vt[((size_t)b * DMODEL + nl) * S_LEN + s] = (__bf16)(acc[r] + bias);
    }
  } else {
    const float scl = (region == 0) ? 0.044194173824159216f : 1.0f;  // 1/sqrt(512)
    __bf16* dst = (region == 0) ? Q : Kg;
#pragma unroll
    for (int r = 0; r < 8; ++r)
      dst[(size_t)(m0 + mh + r) * DMODEL + nl] = (__bf16)((acc[r] + bias) * scl);
  }
}

// ---- Kernel 2: attention for one (batch, 16-query row-block) per block ----
// 512 threads = 16 waves. Dynamic LDS:
//   sc[16][4096] fp32 scores (256KB) | qt[16][512] bf16 (16KB) | reductions.
__global__ __launch_bounds__(ATTN_THREADS) void attn_kernel(
    const __bf16* __restrict__ Q, const __bf16* __restrict__ K,
    const __bf16* __restrict__ Vt, const int* __restrict__ mask,
    __bf16* __restrict__ AO) {
  extern __shared__ char smem_raw[];
  float*  sc     = (float*)smem_raw;             // [16][S_LEN]
  __bf16* qt     = (__bf16*)(sc + 16 * S_LEN);   // [16][DMODEL]
  float*  red    = (float*)(qt + 16 * DMODEL);   // [16][32]
  float*  rowmax = red + 16 * 32;                // [16]
  float*  rowsum = rowmax + 16;                  // [16]

  const int tid = threadIdx.x;
  const int bidx = blockIdx.y;
  const int q0 = blockIdx.x * 16;
  const size_t rowbase = (size_t)bidx * S_LEN;
  const int wave = tid >> 5, lane = tid & 31;

  // Stage (pre-scaled) Q tile with one TDM descriptor: 16x512 bf16 tile,
  // row stride 512, moved HBM -> LDS by the Tensor Data Mover. Tensor ops
  // are per-wave (EXEC ignored), so only wave 0 issues and waits; the
  // workgroup barrier publishes the LDS to all waves.
  if (wave == 0) {
    tdm_load_2d_bf16(lds_offset(qt), Q + (rowbase + q0) * DMODEL,
                     /*tensor_d0=*/DMODEL, /*tensor_d1=*/16,
                     /*tile_d0=*/DMODEL, /*tile_d1=*/16, /*stride0=*/DMODEL);
    wait_tensor0();
  }
  __syncthreads();

  const int nloc = lane & 15;
  const int kb = (lane >> 4) << 4;
  const int mh = (lane >> 4) << 3;

  // Phase 1: S = (Q*scale) @ K^T, masked, into LDS. K is row-major so the
  // B-matrix (K^T) column for key (ks+nloc) is lane-contiguous.
  for (int jt = wave; jt < S_LEN / 16; jt += ATTN_THREADS / 32) {
    const int ks = jt * 16;
    const __bf16* kcol = K + (rowbase + ks + nloc) * DMODEL + kb;
    if (jt + ATTN_THREADS / 32 < S_LEN / 16)
      __builtin_prefetch(kcol + (ATTN_THREADS / 32) * 16 * DMODEL, 0, 1);
    v8f acc = {};
#pragma unroll 4
    for (int kc = 0; kc < DMODEL / 32; ++kc) {
      v16bf a  = pack_a_bf16(qt + kc * 32, DMODEL, lane);
      v16bf bm = pack_b_bf16(kcol + kc * 32);
      acc = wmma_bf16(a, bm, acc);
    }
    const int key = ks + nloc;
    const bool live = mask[rowbase + key] != 0;
#pragma unroll
    for (int r = 0; r < 8; ++r)
      sc[(mh + r) * S_LEN + key] = live ? acc[r] : -1000.0f;
  }
  __syncthreads();

  // Phase 2: row max + exp + row sum (P stays unnormalized in LDS).
  {
    const int r = tid & 15, g = tid >> 4;  // 32 column groups of 128
    const int c0 = g * (S_LEN / 32);
    float mx = -3.4e38f;
    for (int c = 0; c < S_LEN / 32; ++c) mx = fmaxf(mx, sc[r * S_LEN + c0 + c]);
    red[r * 32 + g] = mx;
    __syncthreads();
    if (tid < 16) {
      float m2 = red[tid * 32];
      for (int i = 1; i < 32; ++i) m2 = fmaxf(m2, red[tid * 32 + i]);
      rowmax[tid] = m2;
    }
    __syncthreads();
    const float rm = rowmax[r];
    float s = 0.f;
    for (int c = 0; c < S_LEN / 32; ++c) {
      float p = __expf(sc[r * S_LEN + c0 + c] - rm);
      sc[r * S_LEN + c0 + c] = p;
      s += p;
    }
    red[r * 32 + g] = s;
    __syncthreads();
    if (tid < 16) {
      float s2 = 0.f;
      for (int i = 0; i < 32; ++i) s2 += red[tid * 32 + i];
      rowsum[tid] = s2;
    }
  }
  __syncthreads();

  // Phase 3: O = P @ V via Vt[b][d][s]: B column (head-dim n0+nloc) is
  // lane-contiguous along keys. Normalize by row sum in epilogue.
  for (int nt = wave; nt < DMODEL / 16; nt += ATTN_THREADS / 32) {
    const int n0 = nt * 16;
    const __bf16* vcol = Vt + ((size_t)bidx * DMODEL + n0 + nloc) * S_LEN + kb;
    v8f acc = {};
    for (int kc = 0; kc < S_LEN / 32; ++kc) {
      __builtin_prefetch(vcol + kc * 32 + 512, 0, 1);  // stream ahead
      v16bf a  = pack_a_f32(sc + kc * 32, S_LEN, lane);
      v16bf bm = pack_b_bf16(vcol + kc * 32);
      acc = wmma_bf16(a, bm, acc);
    }
#pragma unroll
    for (int r = 0; r < 8; ++r) {
      const int m = mh + r;
      AO[(rowbase + q0 + m) * DMODEL + n0 + nloc] = (__bf16)(acc[r] / rowsum[m]);
    }
  }
}

// ---- Kernel 3: out = AO @ Wo + bo (fp32 out), WoT pre-transposed bf16 ----
__global__ __launch_bounds__(256) void out_proj_kernel(
    const __bf16* __restrict__ AO, const __bf16* __restrict__ WoT,
    const float* __restrict__ bo, float* __restrict__ out) {
  __shared__ __bf16 atile[16 * DMODEL];
  const int tid = threadIdx.x;
  const int m0 = blockIdx.x * 16;
  // AO tile is contiguous bf16: stage straight into LDS with async copies.
  {
    const __bf16* ga = AO + (size_t)m0 * DMODEL;
    const unsigned at_off = lds_offset(atile);
    for (int i = tid * 8; i < 16 * DMODEL; i += 256 * 8)
      async_copy16(at_off + i * 2, ga + i);
    wait_async0();
  }
  __syncthreads();

  const int wave = tid >> 5, lane = tid & 31;
  const int n0 = (blockIdx.y * 8 + wave) * 16;
  const int ncol = n0 + (lane & 15);
  const int kb = (lane >> 4) << 4;
  const int mh = (lane >> 4) << 3;
  const __bf16* bcol = WoT + (size_t)ncol * DMODEL + kb;

  v8f acc = {};
#pragma unroll 4
  for (int kc = 0; kc < DMODEL / 32; ++kc) {
    v16bf a = pack_a_bf16(atile + kc * 32, DMODEL, lane);
    v16bf b = pack_b_bf16(bcol + kc * 32);
    acc = wmma_bf16(a, b, acc);
  }
  const float bias = bo[ncol];
#pragma unroll
  for (int r = 0; r < 8; ++r)
    out[(size_t)(m0 + mh + r) * DMODEL + ncol] = acc[r] + bias;
}

extern "C" void kernel_launch(void* const* d_in, const int* in_sizes, int n_in,
                              void* d_out, int out_size, void* d_ws, size_t ws_size,
                              hipStream_t stream) {
  const float* x    = (const float*)d_in[0];
  const int*   mask = (const int*)d_in[1];
  const float* Wqkv = (const float*)d_in[2];
  const float* bqkv = (const float*)d_in[3];
  const float* Wo   = (const float*)d_in[4];
  const float* bo   = (const float*)d_in[5];
  float* out = (float*)d_out;

  const size_t NTOK = 4 * (size_t)S_LEN;  // B*S = 16384 rows
  const size_t NQ = NTOK * DMODEL;        // 8388608 elems (16MB bf16)
  __bf16* Q     = (__bf16*)d_ws;          // ws: 4*16MB + 2MB = ~66MB
  __bf16* K     = Q + NQ;
  __bf16* Vt    = K + NQ;                 // [B][DMODEL][S_LEN]
  __bf16* AO    = Vt + NQ;
  __bf16* WqkvT = AO + NQ;                // [ND3][DMODEL]
  __bf16* WoT   = WqkvT + (size_t)ND3 * DMODEL;  // [DMODEL][DMODEL]

  prep_weights_kernel<<<(ND3 * DMODEL) / 256, 256, 0, stream>>>(Wqkv, Wo, WqkvT, WoT);

  qkv_proj_kernel<<<dim3((unsigned)(NTOK / 16), ND3 / 128), 256, 0, stream>>>(
      x, WqkvT, bqkv, Q, K, Vt);

  const size_t smem = (size_t)16 * S_LEN * 4 + 16 * DMODEL * 2 + (16 * 32 + 32) * 4;
  (void)hipFuncSetAttribute((const void*)attn_kernel,
                            hipFuncAttributeMaxDynamicSharedMemorySize, (int)smem);
  attn_kernel<<<dim3(S_LEN / 16, 4), ATTN_THREADS, smem, stream>>>(Q, K, Vt, mask, AO);

  out_proj_kernel<<<dim3((unsigned)(NTOK / 16), DMODEL / 128), 256, 0, stream>>>(
      AO, WoT, bo, out);
}
